// WordAttention_41162966565348
// MI455X (gfx1250) — matrix operand order repaired and checked by reading, not compiled, not run on hardware
//
#include <hip/hip_runtime.h>
#include <hip/hip_bf16.h>
#include <stdint.h>
#include <math.h>

// ---------------------------------------------------------------------------
// WordAttention fused pipeline for MI455X (gfx1250, wave32, WMMA).
// Exploits upsample4 structure: cosine over 32x32 maps == cosine over 8x8
// maps (scaled), so attention runs on 64 query rows/n instead of 1024.
// Main GEMM A[64x1024]@V^T[1024x256] per n uses v_wmma_f32_16x16x32_f16 with
// explicit double-buffered A/B fragments (loads overlap WMMA issue).
// ---------------------------------------------------------------------------

typedef _Float16 v16h __attribute__((ext_vector_type(16)));
typedef float    v8f  __attribute__((ext_vector_type(8)));

struct alignas(16) F4 { float x, y, z, w; };
union HalfVec { v16h h; F4 f[2]; };

#define NLB 60          // L*b = 15*4
#define ASTRIDE 1032    // halves per A row in LDS (pad -> conflict-free frags)
#define SMEM_ATTN (64*ASTRIDE*2 + 8*1024*4 + 64*8*4 + 256)

// ---------------- conv1x1 key: K[i][8][1024] ----------------
__global__ void k_key(const float* __restrict__ img, const float* __restrict__ kW,
                      const float* __restrict__ kb, float* __restrict__ K) {
    int t = blockIdx.x * 256 + threadIdx.x;
    if (t >= 4 * 8 * 1024) return;
    int p = t & 1023, kc = (t >> 10) & 7, i = t >> 13;
    const float* ip = img + (size_t)i * 262144 + p;
    const float* wp = kW + kc * 256;
    float acc = kb[kc];
    for (int ci = 0; ci < 256; ++ci) acc = fmaf(wp[ci], ip[(size_t)ci * 1024], acc);
    K[t] = acc;
}

// -------- conv1x1 value -> f16, pre-swizzled into WMMA-B fragment order ----
// Vsw[i][kt][nt][lane][j] = V^T[p][c], kt=p/32, nt=c/16,
// lane=(c%16)+16*((p%32)/16), j=p%16  -> lane-contiguous 32B fragment loads.
__global__ void k_value(const float* __restrict__ img, const float* __restrict__ vW,
                        const float* __restrict__ vb, _Float16* __restrict__ Vsw) {
    int t = blockIdx.x * 256 + threadIdx.x;
    if (t >= 4 * 256 * 1024) return;
    int p = t & 1023, c = (t >> 10) & 255, i = t >> 18;
    const float* ip = img + (size_t)i * 262144 + p;
    const float* wp = vW + (size_t)c * 256;
    float acc = vb[c];
    for (int ci = 0; ci < 256; ++ci) acc = fmaf(wp[ci], ip[(size_t)ci * 1024], acc);
    int kt = p >> 5, pr = p & 31;
    int lane = (c & 15) + ((pr >> 4) << 4);
    int j = pr & 15, nt = c >> 4;
    size_t off = ((size_t)((i * 32 + kt) * 16 + nt)) * 512 + (size_t)lane * 16 + j;
    Vsw[off] = (_Float16)acc;
}

// ---------------- text branch stage 1: ws @ W1 + b1 ----------------
// ws[n][e] = word_seq[n%4][n/4][e];  W1: [768][Cout][16]
__global__ void k_text1(const float* __restrict__ wseq, const float* __restrict__ W1,
                        const float* __restrict__ b1, float* __restrict__ out, int Cout) {
    int t = blockIdx.x * 256 + threadIdx.x;
    if (t >= NLB * Cout * 16) return;
    int pix = t & 15, oc = (t >> 4) % Cout, n = t / (Cout * 16);
    const float* ws = wseq + ((size_t)(n & 3) * 15 + (n >> 2)) * 768;
    float acc = b1[oc];
    for (int e = 0; e < 768; ++e)
        acc = fmaf(ws[e], W1[((size_t)e * Cout + oc) * 16 + pix], acc);
    out[t] = acc;
}

// ---------------- BN batch stats (training mode, biased var) ----------------
__global__ void k_bnstats(const float* __restrict__ y1q, const float* __restrict__ y1v,
                          const float* __restrict__ qg, const float* __restrict__ qb,
                          const float* __restrict__ vg, const float* __restrict__ vb,
                          float* __restrict__ bns) {
    __shared__ float ss[256], sq[256];
    int blk = blockIdx.x, tid = threadIdx.x;
    bool isq = blk < 8;
    int ch = isq ? blk : blk - 8;
    int Cc = isq ? 8 : 64;
    const float* src = isq ? y1q : y1v;
    float s = 0.f, q2 = 0.f;
    for (int idx = tid; idx < 960; idx += 256) {
        int n = idx >> 4, pix = idx & 15;
        float v = src[((size_t)n * Cc + ch) * 16 + pix];
        s += v; q2 += v * v;
    }
    ss[tid] = s; sq[tid] = q2;
    __syncthreads();
    for (int o = 128; o; o >>= 1) {
        if (tid < o) { ss[tid] += ss[tid + o]; sq[tid] += sq[tid + o]; }
        __syncthreads();
    }
    if (tid == 0) {
        float mean = ss[0] / 960.f;
        float var  = sq[0] / 960.f - mean * mean;
        float g  = isq ? qg[ch] : vg[ch];
        float be = isq ? qb[ch] : vb[ch];
        float sc = g * rsqrtf(var + 1e-5f);
        bns[(isq ? 0 : 16) + ch] = sc;
        bns[(isq ? 8 : 80) + ch] = be - mean * sc;
    }
}

// ---------------- BN affine + ReLU ----------------
__global__ void k_bnapply(const float* __restrict__ y1q, const float* __restrict__ y1v,
                          const float* __restrict__ bns, float* __restrict__ aq,
                          float* __restrict__ av) {
    int t = blockIdx.x * 256 + threadIdx.x;
    if (t >= 7680 + 61440) return;
    if (t < 7680) {
        int ch = (t >> 4) & 7;
        aq[t] = fmaxf(fmaf(y1q[t], bns[ch], bns[8 + ch]), 0.f);
    } else {
        int t2 = t - 7680;
        int ch = (t2 >> 4) & 63;
        av[t2] = fmaxf(fmaf(y1v[t2], bns[16 + ch], bns[80 + ch]), 0.f);
    }
}

// -------- ConvTranspose2d(k=4,s=2,p=1) 4x4 -> 8x8, q branch (8->8 ch) ------
// out layout: Qs[n][pos=64][co=8]  (rows of Q for the S GEMM)
__global__ void k_convt_q(const float* __restrict__ aq, const float* __restrict__ W2,
                          const float* __restrict__ b2, float* __restrict__ Qs) {
    int t = blockIdx.x * 256 + threadIdx.x;
    if (t >= NLB * 8 * 64) return;
    int co = t & 7, pos = (t >> 3) & 63, n = t >> 9;
    int y = pos >> 3, x = pos & 7;
    float acc = b2[co];
    for (int ky = 0; ky < 4; ++ky) {
        int ty = y + 1 - ky;
        if (ty < 0 || (ty & 1)) continue;
        int iy = ty >> 1; if (iy > 3) continue;
        for (int kx = 0; kx < 4; ++kx) {
            int tx = x + 1 - kx;
            if (tx < 0 || (tx & 1)) continue;
            int ix = tx >> 1; if (ix > 3) continue;
            for (int ci = 0; ci < 8; ++ci)
                acc = fmaf(aq[((size_t)n * 8 + ci) * 16 + iy * 4 + ix],
                           W2[((size_t)ci * 8 + co) * 16 + ky * 4 + kx], acc);
        }
    }
    Qs[((size_t)n * 64 + pos) * 8 + co] = acc;
}

// -------- ConvTranspose2d, v branch (64->256 ch) -> vts[n][co][pos] --------
__global__ void k_convt_v(const float* __restrict__ av, const float* __restrict__ W2,
                          const float* __restrict__ b2, float* __restrict__ vts) {
    int t = blockIdx.x * 256 + threadIdx.x;
    if (t >= NLB * 256 * 64) return;
    int pos = t & 63, co = (t >> 6) & 255, n = t >> 14;
    int y = pos >> 3, x = pos & 7;
    int kys[2], iys[2], nky = 0, kxs[2], ixs[2], nkx = 0;
    for (int ky = 0; ky < 4; ++ky) {
        int ty = y + 1 - ky;
        if (ty >= 0 && !(ty & 1) && (ty >> 1) <= 3) { kys[nky] = ky; iys[nky] = ty >> 1; ++nky; }
    }
    for (int kx = 0; kx < 4; ++kx) {
        int tx = x + 1 - kx;
        if (tx >= 0 && !(tx & 1) && (tx >> 1) <= 3) { kxs[nkx] = kx; ixs[nkx] = tx >> 1; ++nkx; }
    }
    float acc = b2[co];
    for (int ci = 0; ci < 64; ++ci) {
        const float* ain = av + ((size_t)n * 64 + ci) * 16;
        const float* wr  = W2 + ((size_t)ci * 256 + co) * 16;
        for (int a = 0; a < nky; ++a)
            for (int bb = 0; bb < nkx; ++bb)
                acc = fmaf(ain[iys[a] * 4 + ixs[bb]], wr[kys[a] * 4 + kxs[bb]], acc);
    }
    vts[((size_t)n * 256 + co) * 64 + pos] = acc;
}

// ---------------- per-n sum of squares of v_txt_small ----------------
__global__ void k_s2(const float* __restrict__ vts, float* __restrict__ s2n) {
    __shared__ float buf[256];
    int n = blockIdx.x, tid = threadIdx.x;
    const float* src = vts + (size_t)n * 16384;
    float s = 0.f;
    for (int idx = tid; idx < 16384; idx += 256) { float v = src[idx]; s = fmaf(v, v, s); }
    buf[tid] = s;
    __syncthreads();
    for (int o = 128; o; o >>= 1) {
        if (tid < o) buf[tid] += buf[tid + o];
        __syncthreads();
    }
    if (tid == 0) s2n[n] = buf[0];
}

// ---- fragment loaders for the WMMA loop (A from LDS, B from swizzled glb) --
__device__ __forceinline__ void load_frags(const _Float16* __restrict__ Alds,
                                           int mrow, int klo,
                                           const _Float16* __restrict__ Vb,
                                           int lane, int nh, int kt,
                                           HalfVec& ua, HalfVec ub[8]) {
    const _Float16* ap = Alds + (size_t)mrow * ASTRIDE + kt * 32 + klo;
    ua.f[0] = *(const F4*)ap;
    ua.f[1] = *(const F4*)(ap + 8);
#pragma unroll
    for (int t = 0; t < 8; ++t) {
        const _Float16* bp = Vb + ((size_t)(kt * 16 + nh * 8 + t)) * 512 + (size_t)lane * 16;
        ub[t].f[0] = *(const F4*)bp;
        ub[t].f[1] = *(const F4*)(bp + 8);
    }
}

__device__ __forceinline__ void wmma8(const HalfVec& ua, const HalfVec ub[8], v8f acc[8]) {
#pragma unroll
    for (int t = 0; t < 8; ++t)
        acc[t] = __builtin_amdgcn_wmma_f32_16x16x32_f16(
            false, ua.h, false, ub[t].h, (short)0, acc[t], false, false);
}

// ---------------- fused attention: S -> softmax -> A@V^T (WMMA) -> cosine
// partials.  One workgroup per n (60 blocks, 8 waves, ~167KB LDS, so exactly
// one workgroup per WGP: registers are cheap, occupancy is LDS-bound).
__global__ __launch_bounds__(256, 1) void k_attn(const float* __restrict__ Kbuf,
                                                 const float* __restrict__ Qs,
                                                 const _Float16* __restrict__ Vsw,
                                                 const float* __restrict__ vts,
                                                 float* __restrict__ dotn,
                                                 float* __restrict__ s1n) {
    extern __shared__ char smem[];
    _Float16* Alds = (_Float16*)smem;                              // 64 x 1032 f16
    float* Klds = (float*)(smem + 64 * ASTRIDE * 2);               // 8 x 1024 f32
    float* Qlds = Klds + 8 * 1024;                                 // 64 x 8 f32
    float* red  = Qlds + 64 * 8;

    int n = blockIdx.x, ib = n & 3;
    int tid = threadIdx.x, w = tid >> 5, lane = tid & 31;

    const float* Kg = Kbuf + (size_t)ib * 8192;
    for (int j = tid; j < 8192; j += 256) Klds[j] = Kg[j];
    const float* Qg = Qs + (size_t)n * 512;
    for (int j = tid; j < 512; j += 256) Qlds[j] = Qg[j];
    __syncthreads();

    // ---- softmax rows: wave w owns rows q = w*8 .. w*8+7, S kept in regs ----
    for (int r = 0; r < 8; ++r) {
        int q = w * 8 + r;
        float qv[8];
#pragma unroll
        for (int k = 0; k < 8; ++k) qv[k] = Qlds[q * 8 + k];
        float sv[32];
        float mx = -3.4e38f;
#pragma unroll
        for (int j = 0; j < 32; ++j) {
            int p = j * 32 + lane;
            float s = 0.f;
#pragma unroll
            for (int k = 0; k < 8; ++k) s = fmaf(qv[k], Klds[k * 1024 + p], s);
            sv[j] = s;
            mx = fmaxf(mx, s);
        }
        for (int off = 16; off; off >>= 1) mx = fmaxf(mx, __shfl_xor(mx, off, 32));
        float sum = 0.f;
#pragma unroll
        for (int j = 0; j < 32; ++j) { float e = expf(sv[j] - mx); sv[j] = e; sum += e; }
        for (int off = 16; off; off >>= 1) sum += __shfl_xor(sum, off, 32);
        float inv = 1.f / sum;
        _Float16* arow = Alds + (size_t)q * ASTRIDE;
#pragma unroll
        for (int j = 0; j < 32; ++j) arow[j * 32 + lane] = (_Float16)(sv[j] * inv);
    }
    __syncthreads();

    // ---- O = A @ V^T via v_wmma_f32_16x16x32_f16, double-buffered frags ----
    // wave w: mt = w/2 (M tile of 4), nh = w&1 (half of 16 N tiles)
    int mt = w >> 1, nh = w & 1;
    v8f acc[8];
#pragma unroll
    for (int t = 0; t < 8; ++t)
#pragma unroll
        for (int r = 0; r < 8; ++r) acc[t][r] = 0.f;

    int mrow = mt * 16 + (lane & 15);
    int klo  = (lane >> 4) * 16;                // A frag: halves K{lo..lo+15}
    const _Float16* Vb = Vsw + (size_t)ib * 32 * 16 * 512;

    HalfVec uaA, uaB;
    HalfVec ubA[8], ubB[8];
    load_frags(Alds, mrow, klo, Vb, lane, nh, 0, uaA, ubA);
    for (int kt = 0; kt < 32; kt += 2) {
        int ktp = kt + 4 < 32 ? kt + 4 : kt;    // prefetch distance 4
        __builtin_prefetch(Vb + ((size_t)(ktp * 16 + nh * 8)) * 512 + (size_t)lane * 16, 0, 1);
        load_frags(Alds, mrow, klo, Vb, lane, nh, kt + 1, uaB, ubB);
        wmma8(uaA, ubA, acc);                   // consumes buffer A while B lands
        if (kt + 2 < 32)
            load_frags(Alds, mrow, klo, Vb, lane, nh, kt + 2, uaA, ubA);
        wmma8(uaB, ubB, acc);                   // consumes buffer B while A lands
    }

    // ---- fused cosine partials: dot(img_att, v_txt) and ||img_att||^2 ----
    float dot = 0.f, s1 = 0.f;
    int q0 = mt * 16 + (lane >> 4) * 8;         // D frag: M = r + 8*(lane/16)
#pragma unroll
    for (int t = 0; t < 8; ++t) {
        int c = (nh * 8 + t) * 16 + (lane & 15);
        const float* vp = vts + ((size_t)n * 256 + c) * 64 + q0;
        F4 v0 = *(const F4*)vp;
        F4 v1 = *(const F4*)(vp + 4);
        float vv[8] = {v0.x, v0.y, v0.z, v0.w, v1.x, v1.y, v1.z, v1.w};
#pragma unroll
        for (int r = 0; r < 8; ++r) {
            float o = acc[t][r];
            dot = fmaf(o, vv[r], dot);
            s1  = fmaf(o, o, s1);
        }
    }
    for (int off = 16; off; off >>= 1) {
        dot += __shfl_xor(dot, off, 32);
        s1  += __shfl_xor(s1,  off, 32);
    }
    if (lane == 0) { red[w] = dot; red[8 + w] = s1; }
    __syncthreads();
    if (tid == 0) {
        float d = 0.f, s = 0.f;
        for (int k = 0; k < 8; ++k) { d += red[k]; s += red[8 + k]; }
        dotn[n] = d;
        s1n[n] = s;
    }
}

// ---------------- final cosine over chunks n in [15i, 15i+15) ----------------
__global__ void k_final(const float* __restrict__ dotn, const float* __restrict__ s1n,
                        const float* __restrict__ s2n, float* __restrict__ out) {
    int i = threadIdx.x;
    if (i < 4) {
        float d = 0.f, a = 0.f, b = 0.f;
        for (int k = 0; k < 15; ++k) {
            int n = i * 15 + k;
            d += dotn[n]; a += s1n[n]; b += s2n[n];
        }
        // full-res: numerator = 16*d, denom = 4*sqrt(a) * 4*sqrt(b)
        float denom = fmaxf(16.f * sqrtf(a * b), 1e-8f);
        out[i] = (16.f * d) / denom;
    }
}

extern "C" void kernel_launch(void* const* d_in, const int* in_sizes, int n_in,
                              void* d_out, int out_size, void* d_ws, size_t ws_size,
                              hipStream_t stream) {
    (void)in_sizes; (void)n_in; (void)out_size; (void)ws_size;
    const float* img    = (const float*)d_in[0];
    const float* wseq   = (const float*)d_in[1];
    const float* key_W  = (const float*)d_in[2];
    const float* key_b  = (const float*)d_in[3];
    const float* vimg_W = (const float*)d_in[4];
    const float* vimg_b = (const float*)d_in[5];
    const float* qW1    = (const float*)d_in[6];
    const float* qb1    = (const float*)d_in[7];
    const float* q_g    = (const float*)d_in[8];
    const float* q_be   = (const float*)d_in[9];
    const float* qW2    = (const float*)d_in[10];
    const float* qb2    = (const float*)d_in[11];
    const float* vW1    = (const float*)d_in[12];
    const float* vb1    = (const float*)d_in[13];
    const float* v_g    = (const float*)d_in[14];
    const float* v_be   = (const float*)d_in[15];
    const float* vW2    = (const float*)d_in[16];
    const float* vb2    = (const float*)d_in[17];
    float* out = (float*)d_out;

    char* base = (char*)d_ws;
    size_t off = 0;
    auto alloc = [&](size_t bytes) -> void* {
        void* p = base + off;
        off = (off + bytes + 255) & ~(size_t)255;
        return p;
    };
    float*    Kbuf = (float*)alloc(4 * 8 * 1024 * 4);        // 128 KB
    _Float16* Vsw  = (_Float16*)alloc(4 * 1024 * 256 * 2);   // 2 MB
    float*    y1q  = (float*)alloc(NLB * 128 * 4);
    float*    y1v  = (float*)alloc(NLB * 1024 * 4);
    float*    aq   = (float*)alloc(NLB * 128 * 4);
    float*    av   = (float*)alloc(NLB * 1024 * 4);
    float*    bns  = (float*)alloc(144 * 4);
    float*    Qsb  = (float*)alloc(NLB * 64 * 8 * 4);
    float*    vts  = (float*)alloc((size_t)NLB * 256 * 64 * 4); // 3.75 MB
    float*    s2n  = (float*)alloc(NLB * 4);
    float*    dotn = (float*)alloc(NLB * 4);
    float*    s1n  = (float*)alloc(NLB * 4);

    k_key   <<<128,  256, 0, stream>>>(img, key_W, key_b, Kbuf);
    k_value <<<4096, 256, 0, stream>>>(img, vimg_W, vimg_b, Vsw);
    k_text1 <<<30,   256, 0, stream>>>(wseq, qW1, qb1, y1q, 8);
    k_text1 <<<240,  256, 0, stream>>>(wseq, vW1, vb1, y1v, 64);
    k_bnstats<<<72,  256, 0, stream>>>(y1q, y1v, q_g, q_be, v_g, v_be, bns);
    k_bnapply<<<270, 256, 0, stream>>>(y1q, y1v, bns, aq, av);
    k_convt_q<<<120, 256, 0, stream>>>(aq, qW2, qb2, Qsb);
    k_convt_v<<<3840,256, 0, stream>>>(av, vW2, vb2, vts);
    k_s2     <<<60,  256, 0, stream>>>(vts, s2n);
    k_attn   <<<NLB, 256, SMEM_ATTN, stream>>>(Kbuf, Qsb, Vsw, vts, dotn, s1n);
    k_final  <<<1,   32, 0, stream>>>(dotn, s1n, s2n, out);
}
